// BarrierNet_48266842472810
// MI455X (gfx1250) — matrix operand order repaired and checked by reading, not compiled, not run on hardware
//
#include <hip/hip_runtime.h>
#include <hip/hip_bf16.h>
#include <math.h>

// ---------------------------------------------------------------------------
// BarrierNet fused: 3-layer MLP on v_wmma_f32_16x16x32_f16 + per-row 2D QP.
// Weights staged per-block into LDS (double buffered) via async-to-LDS copies;
// B tiles software-pipelined (load n+1 while WMMA n executes).
// ---------------------------------------------------------------------------

typedef __attribute__((ext_vector_type(16))) _Float16 v16h;
typedef __attribute__((ext_vector_type(8)))  _Float16 v8h;
typedef __attribute__((ext_vector_type(8)))  float    v8f;
typedef int v4i_ __attribute__((vector_size(16)));

#define NB       32768
#define NFEAT    66
#define KPAD1    96      // 66 padded to multiple of 32
#define H        256
#define NCON     10
#define ROWS     128     // batch rows per block (8 waves x 16 rows)
#define QP_ITERS 30
#define SAFE2    0.64f   // SAFE_DIST^2

#if __has_builtin(__builtin_amdgcn_global_load_async_to_lds_b128)
#define HAVE_ASYNC_LDS 1
#else
#define HAVE_ASYNC_LDS 0
#endif

// 16-byte global->LDS copy (async when available).
__device__ __forceinline__ void cp16_to_lds(_Float16* lds_dst, const _Float16* g_src) {
#if HAVE_ASYNC_LDS
  __builtin_amdgcn_global_load_async_to_lds_b128(
      (__attribute__((address_space(1))) v4i_*)(g_src),
      (__attribute__((address_space(3))) v4i_*)(lds_dst), 0, 0);
#else
  *(v8h*)lds_dst = *(const v8h*)g_src;
#endif
}

__device__ __forceinline__ void async_copies_done() {
#if HAVE_ASYNC_LDS
#if __has_builtin(__builtin_amdgcn_s_wait_asynccnt)
  __builtin_amdgcn_s_wait_asynccnt(0);
#else
  asm volatile("s_wait_asynccnt 0" ::: "memory");
#endif
#endif
}

// ------------------------- weight f32 -> f16 prep --------------------------
__global__ __launch_bounds__(256) void prep_weights_f16(
    const float* __restrict__ W1, const float* __restrict__ W21,
    const float* __restrict__ W22,
    _Float16* __restrict__ W1h, _Float16* __restrict__ W21h,
    _Float16* __restrict__ W22h)
{
  int i = blockIdx.x * blockDim.x + threadIdx.x;
  if (i < H * KPAD1) {
    int r = i / KPAD1, c = i - r * KPAD1;
    W1h[i] = (c < NFEAT) ? (_Float16)W1[r * NFEAT + c] : (_Float16)0.f;
  }
  if (i < H * H) {
    W21h[i] = (_Float16)W21[i];
    W22h[i] = (_Float16)W22[i];
  }
}

// A tile (16xK slab, f16) from LDS with CDNA5 A-matrix swizzle:
// lane<16: e<8 -> K=k0+e, e>=8 -> K=k0+e+8 ; lane>=16: +8 on both groups.
__device__ __forceinline__ v16h load_a_tile(const _Float16* base, int lane,
                                            int k0, int stride) {
  int m  = lane & 15;
  int kb = k0 + ((lane >> 4) << 3);
  const _Float16* r = base + m * stride + kb;
  v8h lo = *(const v8h*)(r);        // K = kb .. kb+7
  v8h hi = *(const v8h*)(r + 16);   // K = kb+16 .. kb+23
  return __builtin_shufflevector(lo, hi, 0,1,2,3,4,5,6,7,8,9,10,11,12,13,14,15);
}

// B tile (32x16) from the LDS weight slab laid out [n][32]:
// lane holds column n = n0+(lane&15); e<8 -> K=kb+e, e>=8 -> K=kb+e+8, kb = 8*(lane>>4).
__device__ __forceinline__ v16h load_b_lds(const _Float16* w, int lane, int n0) {
  int n  = n0 + (lane & 15);
  int kb = (lane >> 4) << 3;
  const _Float16* r = w + n * 32 + kb;
  v8h lo = *(const v8h*)(r);
  v8h hi = *(const v8h*)(r + 16);
  return __builtin_shufflevector(lo, hi, 0,1,2,3,4,5,6,7,8,9,10,11,12,13,14,15);
}

// Stage one 256x32 f16 weight k-slab (16 KB) into LDS: 1024 x 16B chunks.
__device__ __forceinline__ void stage_w_slab(_Float16* dst, const _Float16* W,
                                             int k0, int kpad, int tid) {
  #pragma unroll
  for (int i = 0; i < 4; ++i) {
    int c   = tid + i * 256;     // chunk id, 4 chunks per row
    int row = c >> 2;
    int off = (c & 3) * 8;       // halfword offset within row slab
    cp16_to_lds(dst + row * 32 + off, W + row * kpad + k0 + off);
  }
}

// 16 N-tiles of WMMA against one 32-wide K slab, B tiles pipelined one ahead.
__device__ __forceinline__ void wmma_sweep(v8f acc[16], v16h a,
                                           const _Float16* wcur, int lane) {
  v16h b = load_b_lds(wcur, lane, 0);
  #pragma unroll
  for (int n = 0; n < 16; ++n) {
    v16h bcur = b;
    if (n < 15) b = load_b_lds(wcur, lane, (n + 1) * 16);  // prefetch next tile
    acc[n] = __builtin_amdgcn_wmma_f32_16x16x32_f16(
        false, a, false, bcur, (short)0, acc[n], false, false);
  }
}

__global__ __launch_bounds__(256) void barriernet_fused(
    const float* __restrict__ obs,
    const _Float16* __restrict__ W1h, const float* __restrict__ b1,
    const _Float16* __restrict__ W21h, const float* __restrict__ b21,
    const _Float16* __restrict__ W22h, const float* __restrict__ b22,
    const float* __restrict__ W31, const float* __restrict__ b31,
    const float* __restrict__ W32, const float* __restrict__ b32,
    float* __restrict__ out)
{
  __shared__ __align__(32) _Float16 obs_s[ROWS * KPAD1];   // 24 KB
  __shared__ __align__(32) _Float16 x_s[ROWS * H];         // 64 KB
  __shared__ __align__(32) _Float16 w_s[2][H * 32];        // 2 x 16 KB double buffer
  __shared__ float    unom_s[ROWS * 2];
  __shared__ float    alpha_s[ROWS];

  const int tid  = threadIdx.x;
  const int lane = tid & 31;
  const int wave = tid >> 5;
  const int row0 = blockIdx.x * ROWS;
  const int hi8  = (lane >> 4) << 3;   // acc VGPR r maps to row  wave*16 + hi8 + r

  // ---- stage obs -> LDS as f16, K padded 66 -> 96; prefetch W1 slab 0 ----
  stage_w_slab(w_s[0], W1h, 0, KPAD1, tid);
  for (int i = tid; i < ROWS * KPAD1; i += 256) {
    int r = i / KPAD1, c = i - r * KPAD1;
    float v = (c < NFEAT) ? obs[(row0 + r) * NFEAT + c] : 0.f;
    obs_s[i] = (_Float16)v;
  }
  async_copies_done();
  __syncthreads();

  v8f acc[16];

  // =============== layer 1: x = relu(obs @ W1^T + b1) ===============
  #pragma unroll
  for (int n = 0; n < 16; ++n) acc[n] = {};
  #pragma unroll 1
  for (int ks = 0; ks < 3; ++ks) {
    const _Float16* wcur = w_s[ks & 1];
    if (ks < 2) stage_w_slab(w_s[(ks + 1) & 1], W1h, (ks + 1) * 32, KPAD1, tid);
    v16h a = load_a_tile(obs_s + wave * 16 * KPAD1, lane, ks * 32, KPAD1);
    wmma_sweep(acc, a, wcur, lane);
    async_copies_done();
    __syncthreads();
  }
  // bias + relu, store f16 into x_s (C-layout scatter; wave-private rows)
  #pragma unroll
  for (int n = 0; n < 16; ++n) {
    int ncol = n * 16 + (lane & 15);
    float bb = b1[ncol];
    #pragma unroll
    for (int r = 0; r < 8; ++r) {
      float v = acc[n][r] + bb;
      v = v > 0.f ? v : 0.f;
      x_s[(wave * 16 + hi8 + r) * H + ncol] = (_Float16)v;
    }
  }

  // =============== layer 2: x21 = relu(x @ W21^T + b21); unom head ===============
  #pragma unroll
  for (int n = 0; n < 16; ++n) acc[n] = {};
  stage_w_slab(w_s[0], W21h, 0, H, tid);
  async_copies_done();
  __syncthreads();
  #pragma unroll 1
  for (int ks = 0; ks < 8; ++ks) {
    const _Float16* wcur = w_s[ks & 1];
    if (ks < 7) stage_w_slab(w_s[(ks + 1) & 1], W21h, (ks + 1) * 32, H, tid);
    v16h a = load_a_tile(x_s + wave * 16 * H, lane, ks * 32, H);
    wmma_sweep(acc, a, wcur, lane);
    if (ks < 7) { async_copies_done(); __syncthreads(); }
  }
  {
    float p0[8], p1[8];
    #pragma unroll
    for (int r = 0; r < 8; ++r) { p0[r] = 0.f; p1[r] = 0.f; }
    #pragma unroll
    for (int n = 0; n < 16; ++n) {
      int ncol = n * 16 + (lane & 15);
      float bb = b21[ncol];
      float w0 = W31[ncol];
      float w1 = W31[H + ncol];
      #pragma unroll
      for (int r = 0; r < 8; ++r) {
        float v = acc[n][r] + bb;
        v = v > 0.f ? v : 0.f;
        p0[r] += v * w0;
        p1[r] += v * w1;
      }
    }
    #pragma unroll
    for (int m = 1; m < 16; m <<= 1) {
      #pragma unroll
      for (int r = 0; r < 8; ++r) {
        p0[r] += __shfl_xor(p0[r], m, 32);
        p1[r] += __shfl_xor(p1[r], m, 32);
      }
    }
    if ((lane & 15) == 0) {
      #pragma unroll
      for (int r = 0; r < 8; ++r) {
        int rr = wave * 16 + hi8 + r;
        unom_s[rr * 2 + 0] = p0[r] + b31[0];
        unom_s[rr * 2 + 1] = p1[r] + b31[1];
      }
    }
  }

  // =============== layer 3: x22 = relu(x @ W22^T + b22); alpha head ===============
  #pragma unroll
  for (int n = 0; n < 16; ++n) acc[n] = {};
  stage_w_slab(w_s[0], W22h, 0, H, tid);
  async_copies_done();
  __syncthreads();
  #pragma unroll 1
  for (int ks = 0; ks < 8; ++ks) {
    const _Float16* wcur = w_s[ks & 1];
    if (ks < 7) stage_w_slab(w_s[(ks + 1) & 1], W22h, (ks + 1) * 32, H, tid);
    v16h a = load_a_tile(x_s + wave * 16 * H, lane, ks * 32, H);
    wmma_sweep(acc, a, wcur, lane);
    if (ks < 7) { async_copies_done(); __syncthreads(); }
  }
  {
    float p2[8];
    #pragma unroll
    for (int r = 0; r < 8; ++r) p2[r] = 0.f;
    #pragma unroll
    for (int n = 0; n < 16; ++n) {
      int ncol = n * 16 + (lane & 15);
      float bb = b22[ncol];
      float w  = W32[ncol];
      #pragma unroll
      for (int r = 0; r < 8; ++r) {
        float v = acc[n][r] + bb;
        v = v > 0.f ? v : 0.f;
        p2[r] += v * w;
      }
    }
    #pragma unroll
    for (int m = 1; m < 16; m <<= 1) {
      #pragma unroll
      for (int r = 0; r < 8; ++r) p2[r] += __shfl_xor(p2[r], m, 32);
    }
    if ((lane & 15) == 0) {
      #pragma unroll
      for (int r = 0; r < 8; ++r) {
        float z = p2[r] + b32[0];
        alpha_s[wave * 16 + hi8 + r] = 4.f / (1.f + expf(-z));
      }
    }
  }
  __syncthreads();

  // =============== per-row QP (n=2, m=10 interior point) ===============
  if (tid < ROWS) {
    const int row = row0 + tid;
    const float* ob = obs + row * NFEAT;
    const float alpha = alpha_s[tid];

    float G0[NCON], G1[NCON], h[NCON];
    #pragma unroll
    for (int k = 0; k < NCON; ++k) {
      float r0 = ob[6 + 6 * k + 0];
      float r1 = ob[6 + 6 * k + 1];
      float v0 = ob[6 + 6 * k + 2];
      float v1 = ob[6 + 6 * k + 3];
      float mk = ob[6 + 6 * k + 5];
      mk = fminf(fmaxf(mk, 0.f), 1.f);
      float barrier = r0 * r0 + r1 * r1 - SAFE2;
      float lf = -2.f * (r0 * v0 + r1 * v1);
      G0[k] = -2.f * r0 * mk;
      G1[k] = -2.f * r1 * mk;
      h[k]  = (lf + alpha * barrier) * mk;
    }
    const float pp0 = -2.f * unom_s[tid * 2 + 0];
    const float pp1 = -2.f * unom_s[tid * 2 + 1];

    float x0 = 0.f, x1 = 0.f;
    float s[NCON], lam[NCON];
    #pragma unroll
    for (int k = 0; k < NCON; ++k) { s[k] = 1.f; lam[k] = 1.f; }

    #pragma unroll 1
    for (int it = 0; it < QP_ITERS; ++it) {
      float rd0 = 2.f * x0 + pp0, rd1 = 2.f * x1 + pp1;
      float musum = 0.f;
      float rp[NCON];
      #pragma unroll
      for (int k = 0; k < NCON; ++k) {
        rd0 += G0[k] * lam[k];
        rd1 += G1[k] * lam[k];
        rp[k] = G0[k] * x0 + G1[k] * x1 + s[k] - h[k];
        musum += lam[k] * s[k];
      }
      float mu = musum * (1.f / NCON);
      float M00 = 2.f, M01 = 0.f, M11 = 2.f;
      float q0 = -rd0, q1 = -rd1;
      #pragma unroll
      for (int k = 0; k < NCON; ++k) {
        float rc = lam[k] * s[k] - 0.1f * mu;
        float Dk = lam[k] / s[k];
        M00 += G0[k] * Dk * G0[k];
        M01 += G0[k] * Dk * G1[k];
        M11 += G1[k] * Dk * G1[k];
        float t = (lam[k] * rp[k] - rc) / s[k];
        q0 -= G0[k] * t;
        q1 -= G1[k] * t;
      }
      float det = M00 * M11 - M01 * M01;
      float inv = 1.f / det;
      float dx0 = (M11 * q0 - M01 * q1) * inv;
      float dx1 = (M00 * q1 - M01 * q0) * inv;
      float a_s = 1e30f, a_l = 1e30f;
      float dsv[NCON], dlv[NCON];
      #pragma unroll
      for (int k = 0; k < NCON; ++k) {
        float dsk = -rp[k] - (G0[k] * dx0 + G1[k] * dx1);
        float rc  = lam[k] * s[k] - 0.1f * mu;
        float dlk = (-rc - lam[k] * dsk) / s[k];
        dsv[k] = dsk; dlv[k] = dlk;
        if (dsk < 0.f) a_s = fminf(a_s, -s[k] / dsk);
        if (dlk < 0.f) a_l = fminf(a_l, -lam[k] / dlk);
      }
      float a = fminf(1.f, 0.99f * fminf(a_s, a_l));
      x0 += a * dx0; x1 += a * dx1;
      #pragma unroll
      for (int k = 0; k < NCON; ++k) { s[k] += a * dsv[k]; lam[k] += a * dlv[k]; }
    }
    out[row * 2 + 0] = x0;
    out[row * 2 + 1] = x1;
  }
}

extern "C" void kernel_launch(void* const* d_in, const int* in_sizes, int n_in,
                              void* d_out, int out_size, void* d_ws, size_t ws_size,
                              hipStream_t stream) {
  (void)in_sizes; (void)n_in; (void)out_size; (void)ws_size;
  const float* obs = (const float*)d_in[0];
  const float* W1  = (const float*)d_in[1];
  const float* b1  = (const float*)d_in[2];
  const float* W21 = (const float*)d_in[3];
  const float* b21 = (const float*)d_in[4];
  const float* W22 = (const float*)d_in[5];
  const float* b22 = (const float*)d_in[6];
  const float* W31 = (const float*)d_in[7];
  const float* b31 = (const float*)d_in[8];
  const float* W32 = (const float*)d_in[9];
  const float* b32 = (const float*)d_in[10];
  float* out = (float*)d_out;

  // workspace layout: f16 weights (needs ~304 KB)
  _Float16* W1h  = (_Float16*)d_ws;                               // 256*96*2  = 48 KB
  _Float16* W21h = (_Float16*)((char*)d_ws + 49152);              // 256*256*2 = 128 KB
  _Float16* W22h = (_Float16*)((char*)d_ws + 49152 + 131072);     // 128 KB

  prep_weights_f16<<<(H * H + 255) / 256, 256, 0, stream>>>(
      W1, W21, W22, W1h, W21h, W22h);

  barriernet_fused<<<NB / ROWS, 256, 0, stream>>>(
      obs, W1h, b1, W21h, b21, W22h, b22, W31, b31, W32, b32, out);
}
